// TokenizedHLMv9Block_8186207666901
// MI455X (gfx1250) — matrix operand order repaired and checked by reading, not compiled, not run on hardware
//
#include <hip/hip_runtime.h>
#include <hip/hip_bf16.h>
#include <math.h>

typedef __attribute__((ext_vector_type(16))) __bf16 v16bf;
typedef __attribute__((ext_vector_type(8)))  __bf16 v8bf;
typedef __attribute__((ext_vector_type(8)))  float  v8f;

// ---------------------------------------------------------------------------
// WMMA helpers (gfx1250, wave32). D = A(16x32 bf16) * B(32x16 bf16) + C(f32)
// ---------------------------------------------------------------------------
__device__ inline v8f wmma_bf16(v16bf a, v16bf b, v8f c) {
  // 8 args: (neg_a, A, neg_b, B, c_mod, C, reuse_a, reuse_b)
  return __builtin_amdgcn_wmma_f32_16x16x32_bf16(false, a, false, b,
                                                 (short)0, c, false, false);
}

// A-matrix 16x32 (MxK), row-major source with row stride ld (elements).
// ISA layout: lane L: m = L&15, half = L>>4.
// elements 0..7  <-> K = 8*half + 0..7 ; elements 8..15 <-> K = 16+8*half+0..7
__device__ inline v16bf load_a_frag(const __bf16* __restrict__ tile, int ld) {
  int lane = threadIdx.x & 31;
  const __bf16* p = tile + (size_t)(lane & 15) * ld + ((lane >> 4) << 3);
  union { v16bf v; v8bf h[2]; } u;
  u.h[0] = *(const v8bf*)(p);
  u.h[1] = *(const v8bf*)(p + 16);
  return u.v;
}

// B-matrix 32x16 fragment from an LDS-staged tile with row stride rowB bytes.
// ISA layout: lane L: n = L&15, half = L>>4; elements 0..7 <-> K = 8*half + 0..7 ; elements 8..15 <-> K = 16+8*half+0..7
// koff = byte offset of the 32-K step inside the staged chunk.
template <int ROWB>
__device__ inline v16bf load_b_frag_lds(const char* buf, int trow, int koff) {
  int lane = threadIdx.x & 31;
  const __bf16* p = (const __bf16*)(buf + (trow * 16 + (lane & 15)) * ROWB +
                                    koff + ((lane >> 4) * 16));
  union { v16bf v; v8bf h[2]; } u;
  u.h[0] = *(const v8bf*)(p);
  u.h[1] = *(const v8bf*)(p + 16);
  return u.v;
}

// ---------------------------------------------------------------------------
// CDNA5 async global->LDS copy (ASYNCcnt-tracked). One lane moves 16 bytes.
// LDS byte offset = low 32 bits of the generic LDS address (aperture layout).
// ---------------------------------------------------------------------------
__device__ inline void async_copy_b128(const void* gaddr, void* lds_ptr) {
  unsigned ldsoff = (unsigned)(size_t)lds_ptr;
  asm volatile("global_load_async_to_lds_b128 %0, %1, off"
               :: "v"(ldsoff), "v"(gaddr) : "memory");
}
__device__ inline void wait_async0() {
  asm volatile("s_wait_asynccnt 0x0" ::: "memory");
}

// Stage a 64-row x 64-K bf16 chunk of Bt (N x K row-major, stride K) into LDS
// (row stride 128 B). 8 KB total: 256 threads x 32 B = 2 async b128 each.
__device__ inline void stage_b64x64(const __bf16* __restrict__ Bt, int n0,
                                    int k0, int K, char* buf) {
  int t   = threadIdx.x;
  int row = t >> 2;             // 0..63
  int ch  = (t & 3) * 32;       // 32-byte sub-chunk within 128-byte row
  const char* ga = (const char*)(Bt + (size_t)(n0 + row) * K + k0) + ch;
  char* lp = buf + row * 128 + ch;
  async_copy_b128(ga,      lp);
  async_copy_b128(ga + 16, lp + 16);
}

__device__ inline float wred32(float v) {
  #pragma unroll
  for (int o = 16; o > 0; o >>= 1) v += __shfl_xor(v, o, 32);
  return v;
}

__device__ inline float sigmoidf_(float z) { return 1.f / (1.f + __expf(-z)); }

// ---------------------------------------------------------------------------
// GEMM: C(f32, MxN) = A(bf16, MxK) * B (Bt = B^T bf16, NxK), optional += (ADD)
// block = 256 threads (8 waves); wave -> 32x64 C tile (2x4 accums);
// block tile 256(M) x 64(N); grid (M/256, N/64). K % 64 == 0.
// B chunk (64 rows x 64 K, shared by all 8 waves) double-buffered in LDS via
// async copies; one wait+barrier per 64-K chunk (2 WMMA k-steps).
// ---------------------------------------------------------------------------
template <bool ADD>
__global__ __launch_bounds__(256) void k_gemm_t(const __bf16* __restrict__ A,
                                                const __bf16* __restrict__ Bt,
                                                float* __restrict__ C,
                                                int M, int N, int K) {
  __shared__ __align__(64) char Bs[2][64 * 128];  // 2 x 8 KB
  int wave = threadIdx.x >> 5;
  int lane = threadIdx.x & 31;
  int m0 = (blockIdx.x * 8 + wave) * 32;
  int n0 = blockIdx.y * 64;
  const v8f zero = {0.f,0.f,0.f,0.f,0.f,0.f,0.f,0.f};
  v8f acc[2][4] = {{zero, zero, zero, zero}, {zero, zero, zero, zero}};

  int nch = K >> 6;                       // 64-wide K chunks
  stage_b64x64(Bt, n0, 0, K, Bs[0]);
  for (int c = 0; c < nch; ++c) {
    wait_async0();          // our slice of buffer c&1 is in LDS
    __syncthreads();        // everyone's slice is; prior reads of c^1 done
    if (c + 1 < nch) stage_b64x64(Bt, n0, (c + 1) << 6, K, Bs[(c + 1) & 1]);
    const char* bs = Bs[c & 1];
    #pragma unroll
    for (int s = 0; s < 2; ++s) {         // two 32-K WMMA steps per chunk
      int k0 = (c << 6) + (s << 5);
      v16bf a0 = load_a_frag(A + (size_t)m0 * K + k0, K);
      v16bf a1 = load_a_frag(A + (size_t)(m0 + 16) * K + k0, K);
      #pragma unroll
      for (int t = 0; t < 4; ++t) {
        v16bf b = load_b_frag_lds<128>(bs, t, s * 64);
        acc[0][t] = wmma_bf16(a0, b, acc[0][t]);
        acc[1][t] = wmma_bf16(a1, b, acc[1][t]);
      }
    }
  }

  int n = lane & 15;
  #pragma unroll
  for (int h = 0; h < 2; ++h) {
    int mb = m0 + h * 16 + ((lane >> 4) << 3);  // vgpr r -> + r + 8*(lane/16)
    #pragma unroll
    for (int t = 0; t < 4; ++t)
      #pragma unroll
      for (int r = 0; r < 8; ++r) {
        size_t idx = (size_t)(mb + r) * N + (n0 + t * 16 + n);
        if (ADD) C[idx] += acc[h][t][r]; else C[idx] = acc[h][t][r];
      }
  }
}

// ---------------------------------------------------------------------------
// Fused gate/up GEMM + SwiGLU: FF(bf16, Mx512) = silu(H@Wg) * (H@Wu)
// H: bf16 Mx128, GT/UT: bf16 512x128 (= Wg^T / Wu^T). wave -> 16x32 tile.
// grid (M/128, 512/32). Both 32x32 weight tiles async-staged (4 KB total).
// ---------------------------------------------------------------------------
__global__ __launch_bounds__(256) void k_gemm_gu(const __bf16* __restrict__ H,
                                                 const __bf16* __restrict__ GT,
                                                 const __bf16* __restrict__ UT,
                                                 __bf16* __restrict__ FF) {
  const int K = 128, N = 512;
  __shared__ __align__(64) char Gs[2][32 * 64];
  __shared__ __align__(64) char Us[2][32 * 64];
  int wave = threadIdx.x >> 5;
  int lane = threadIdx.x & 31;
  int m0 = (blockIdx.x * 8 + wave) * 16;
  int n0 = blockIdx.y * 32;
  const v8f zero = {0.f,0.f,0.f,0.f,0.f,0.f,0.f,0.f};
  v8f ag[2] = {zero, zero};
  v8f au[2] = {zero, zero};

  // stage: threads 0..127 -> GT tile, 128..255 -> UT tile (16 B each)
  auto stage = [&](int k0, int b) {
    int t   = threadIdx.x;
    int row = (t >> 2) & 31;
    int ch  = t & 3;
    const __bf16* W = (t & 128) ? UT : GT;
    char* buf = (t & 128) ? Us[b] : Gs[b];
    const char* ga = (const char*)(W + (size_t)(n0 + row) * K + k0) + ch * 16;
    async_copy_b128(ga, buf + row * 64 + ch * 16);
  };

  stage(0, 0);
  #pragma unroll
  for (int s = 0; s < 4; ++s) {                 // K/32 = 4 steps
    wait_async0();
    __syncthreads();
    if (s + 1 < 4) stage((s + 1) << 5, (s + 1) & 1);
    v16bf a = load_a_frag(H + (size_t)m0 * K + (s << 5), K);
    #pragma unroll
    for (int t = 0; t < 2; ++t) {
      ag[t] = wmma_bf16(a, load_b_frag_lds<64>(Gs[s & 1], t, 0), ag[t]);
      au[t] = wmma_bf16(a, load_b_frag_lds<64>(Us[s & 1], t, 0), au[t]);
    }
  }

  int n  = lane & 15;
  int mb = m0 + ((lane >> 4) << 3);
  #pragma unroll
  for (int t = 0; t < 2; ++t)
    #pragma unroll
    for (int r = 0; r < 8; ++r) {
      float g = ag[t][r];
      float u = au[t][r];
      float f = g * sigmoidf_(g) * u;   // silu(g) * u
      FF[(size_t)(mb + r) * N + (n0 + t * 16 + n)] = (__bf16)f;
    }
}

// ---------------------------------------------------------------------------
// Geometric product + gated mix + per-blade LayerNorm -> bf16 H
// Cl(3,0) Cayley table: output blade k = sum of 8 signed (i,j) pairs.
// One block per token; wave w handles blade k = w (128 dims / 32 lanes).
// ---------------------------------------------------------------------------
__device__ const signed char GI[64] = {
  0,1,2,3,4,5,6,7,  0,1,2,3,4,5,6,7,  0,1,2,3,4,5,6,7,  0,1,2,3,4,5,6,7,
  0,1,2,3,4,5,6,7,  0,1,2,3,4,5,6,7,  0,1,2,3,4,5,6,7,  0,1,2,3,4,5,6,7 };
__device__ const signed char GJ[64] = {
  0,1,2,3,4,5,6,7,   // k=0
  1,0,4,5,2,3,7,6,   // k=1
  2,4,0,6,1,7,3,5,   // k=2
  3,5,6,0,7,1,2,4,   // k=3
  4,2,1,7,0,6,5,3,   // k=4
  5,3,7,1,6,0,4,2,   // k=5
  6,7,3,2,5,4,0,1,   // k=6
  7,6,5,4,3,2,1,0 }; // k=7
__device__ const signed char GS[64] = {
  1,1,1,1,-1,-1,-1,-1,   // k=0
  1,1,-1,-1,1,1,-1,-1,   // k=1
  1,1,1,-1,-1,1,1,1,     // k=2
  1,1,1,1,-1,-1,-1,-1,   // k=3
  1,1,-1,1,1,-1,1,1,     // k=4
  1,1,-1,-1,1,1,-1,-1,   // k=5
  1,1,1,-1,-1,1,1,1,     // k=6
  1,1,-1,1,1,-1,1,1 };   // k=7

__global__ __launch_bounds__(256) void k_geo_ln(const float* __restrict__ mv,
                                                const float* __restrict__ iw,
                                                const float* __restrict__ gg,
                                                const float* __restrict__ lnw,
                                                const float* __restrict__ lnb,
                                                __bf16* __restrict__ H) {
  __shared__ float X[1024];  // 8 blades x 128 dims for this token
  int tok = blockIdx.x;
  const float* src = mv + (size_t)tok * 1024;
  for (int i = threadIdx.x; i < 1024; i += 256) X[i] = src[i];
  __syncthreads();

  int k = threadIdx.x >> 5;     // wave id == blade
  int lane = threadIdx.x & 31;
  float g = sigmoidf_(gg[0]);

  float swv[8]; int pi[8], pj[8];
  int base = k * 8;
  #pragma unroll
  for (int p = 0; p < 8; ++p) {
    int i = GI[base + p], j = GJ[base + p];
    swv[p] = (float)GS[base + p] * sigmoidf_(iw[i * 8 + j]);
    pi[p] = i * 128; pj[p] = j * 128;
  }

  float mixed[4]; float s = 0.f, ss = 0.f;
  #pragma unroll
  for (int jj = 0; jj < 4; ++jj) {
    int d = lane + 32 * jj;
    float geo = 0.f;
    #pragma unroll
    for (int p = 0; p < 8; ++p) geo += swv[p] * X[pi[p] + d] * X[pj[p] + d];
    float m = g * geo + (1.f - g) * X[k * 128 + d];
    mixed[jj] = m; s += m; ss += m * m;
  }
  s = wred32(s); ss = wred32(ss);
  float mu = s * (1.f / 128.f);
  float var = ss * (1.f / 128.f) - mu * mu;
  float rs = rsqrtf(var + 1e-5f);

  __bf16* out = H + ((size_t)tok * 8 + k) * 128;
  #pragma unroll
  for (int jj = 0; jj < 4; ++jj) {
    int d = lane + 32 * jj;
    float h = (mixed[jj] - mu) * rs * lnw[d] + lnb[d];
    out[d] = (__bf16)h;
  }
}

// ---------------------------------------------------------------------------
// Final LayerNorm over D=1024 per token row
// ---------------------------------------------------------------------------
__global__ __launch_bounds__(256) void k_final_ln(const float* __restrict__ in,
                                                  const float* __restrict__ w,
                                                  const float* __restrict__ b,
                                                  float* __restrict__ out) {
  int row = blockIdx.x;
  const float* p = in + (size_t)row * 1024;
  float s = 0.f, ss = 0.f;
  for (int i = threadIdx.x; i < 1024; i += 256) { float v = p[i]; s += v; ss += v * v; }
  s = wred32(s); ss = wred32(ss);
  __shared__ float S[8], Q[8];
  int wave = threadIdx.x >> 5, lane = threadIdx.x & 31;
  if (lane == 0) { S[wave] = s; Q[wave] = ss; }
  __syncthreads();
  s = 0.f; ss = 0.f;
  #pragma unroll
  for (int i = 0; i < 8; ++i) { s += S[i]; ss += Q[i]; }
  float mu = s * (1.f / 1024.f);
  float rs = rsqrtf(ss * (1.f / 1024.f) - mu * mu + 1e-5f);
  float* o = out + (size_t)row * 1024;
  for (int i = threadIdx.x; i < 1024; i += 256)
    o[i] = (p[i] - mu) * rs * w[i] + b[i];
}

// ---------------------------------------------------------------------------
// fp32 -> bf16 convert, and transpose-convert (for weight B^T layouts)
// ---------------------------------------------------------------------------
__global__ __launch_bounds__(256) void k_conv(const float* __restrict__ in,
                                              __bf16* __restrict__ out, int n) {
  int i = blockIdx.x * 256 + threadIdx.x;
  if (i < n) out[i] = (__bf16)in[i];
}
__global__ __launch_bounds__(256) void k_tconv(const float* __restrict__ in,
                                               __bf16* __restrict__ out,
                                               int K, int N) {
  int idx = blockIdx.x * 256 + threadIdx.x;
  if (idx < K * N) {
    int k = idx / N, n = idx % N;
    out[(size_t)n * K + k] = (__bf16)in[idx];
  }
}

// ---------------------------------------------------------------------------
// Orchestration
// ---------------------------------------------------------------------------
static inline size_t alup(size_t x) { return (x + 255) & ~(size_t)255; }

extern "C" void kernel_launch(void* const* d_in, const int* in_sizes, int n_in,
                              void* d_out, int out_size, void* d_ws, size_t ws_size,
                              hipStream_t stream) {
  (void)in_sizes; (void)n_in; (void)out_size; (void)ws_size;
  const int BT = 8192;          // B*T tokens
  const int D  = 1024;          // d_model = 8 * 128
  const int DB = 128;           // d_blade
  const int FFN = 512;
  const int RB = BT * 8;        // 65536 token-blade rows

  const float* x    = (const float*)d_in[0];
  const float* Win  = (const float*)d_in[1];
  const float* Wout = (const float*)d_in[2];
  const float* on_w = (const float*)d_in[3];
  const float* on_b = (const float*)d_in[4];

  // workspace carve-up
  char* ws = (char*)d_ws;
  size_t off = 0;
  __bf16* xb    = (__bf16*)(ws + off); off += alup((size_t)BT * D * 2);
  __bf16* WinT  = (__bf16*)(ws + off); off += alup((size_t)D * D * 2);
  __bf16* WoutT = (__bf16*)(ws + off); off += alup((size_t)D * D * 2);
  __bf16* WgT[2]; __bf16* WuT[2]; __bf16* WdT[2];
  for (int r = 0; r < 2; ++r) {
    WgT[r] = (__bf16*)(ws + off); off += alup((size_t)DB * FFN * 2);
    WuT[r] = (__bf16*)(ws + off); off += alup((size_t)DB * FFN * 2);
    WdT[r] = (__bf16*)(ws + off); off += alup((size_t)FFN * DB * 2);
  }
  float*  mvf  = (float*) (ws + off); off += alup((size_t)BT * D * 4);
  __bf16* hb   = (__bf16*)(ws + off); off += alup((size_t)RB * DB * 2);
  __bf16* ffb  = (__bf16*)(ws + off); off += alup((size_t)RB * FFN * 2);
  __bf16* mvb  = (__bf16*)(ws + off); off += alup((size_t)BT * D * 2);
  float*  outf = (float*) (ws + off); off += alup((size_t)BT * D * 4);

  // 1) precision conversions / weight transposes
  k_conv <<<(BT * D + 255) / 256, 256, 0, stream>>>(x, xb, BT * D);
  k_tconv<<<(D * D + 255) / 256, 256, 0, stream>>>(Win,  WinT,  D, D);
  k_tconv<<<(D * D + 255) / 256, 256, 0, stream>>>(Wout, WoutT, D, D);
  for (int r = 0; r < 2; ++r) {
    const float* Wg = (const float*)d_in[5 + 7 * r + 4];
    const float* Wu = (const float*)d_in[5 + 7 * r + 5];
    const float* Wd = (const float*)d_in[5 + 7 * r + 6];
    k_tconv<<<(DB * FFN + 255) / 256, 256, 0, stream>>>(Wg, WgT[r], DB, FFN);
    k_tconv<<<(DB * FFN + 255) / 256, 256, 0, stream>>>(Wu, WuT[r], DB, FFN);
    k_tconv<<<(FFN * DB + 255) / 256, 256, 0, stream>>>(Wd, WdT[r], FFN, DB);
  }

  // 2) mv = x @ Win  (8192x1024x1024)
  k_gemm_t<false><<<dim3(BT / 256, D / 64), 256, 0, stream>>>(xb, WinT, mvf, BT, D, D);

  // 3) two geo rounds
  for (int r = 0; r < 2; ++r) {
    const float* iw  = (const float*)d_in[5 + 7 * r + 0];
    const float* gg  = (const float*)d_in[5 + 7 * r + 1];
    const float* lnw = (const float*)d_in[5 + 7 * r + 2];
    const float* lnb = (const float*)d_in[5 + 7 * r + 3];
    k_geo_ln<<<BT, 256, 0, stream>>>(mvf, iw, gg, lnw, lnb, hb);
    k_gemm_gu<<<dim3(RB / 128, FFN / 32), 256, 0, stream>>>(hb, WgT[r], WuT[r], ffb);
    k_gemm_t<true><<<dim3(RB / 256, DB / 64), 256, 0, stream>>>(ffb, WdT[r], mvf,
                                                                RB, DB, FFN);
  }

  // 4) out = mv @ Wout, then LayerNorm
  k_conv<<<(BT * D + 255) / 256, 256, 0, stream>>>(mvf, mvb, BT * D);
  k_gemm_t<false><<<dim3(BT / 256, D / 64), 256, 0, stream>>>(mvb, WoutT, outf, BT, D, D);
  k_final_ln<<<BT, 256, 0, stream>>>(outf, on_w, on_b, (float*)d_out);
}
